// TrackBHead_83288005804296
// MI455X (gfx1250) — matrix operand order repaired and checked by reading, not compiled, not run on hardware
//
#include <hip/hip_runtime.h>
#include <hip/hip_bf16.h>
#include <stdint.h>

// CDNA5 / gfx1250: wave32, WMMA bf16 16x16x32, fp32 accumulate.

typedef __attribute__((ext_vector_type(16))) __bf16 v16bf;
typedef __attribute__((ext_vector_type(8)))  float  v8f;

#define FREAL 4098
#define FPAD  4128

__device__ __forceinline__ unsigned short f2bf(float x) {
  unsigned int u = __float_as_uint(x);
  u += 0x7FFFu + ((u >> 16) & 1u);   // round-to-nearest-even to bf16
  return (unsigned short)(u >> 16);
}

union FragBF { v16bf v; uint4 u[2]; unsigned short s[16]; };

// A-matrix fragment (16x32 bf16, ISA 7.12.2 layout):
// lane (m = lane&15, h = lane>>4): v0..v3 hold K = k0+8h..+7, v4..v7 hold K = k0+16+8h..+7
__device__ __forceinline__ v16bf load_a_frag(const unsigned short* p) {
  FragBF r;
  r.u[0] = *(const uint4*)p;          // K = k0 + 8h .. +7
  r.u[1] = *(const uint4*)(p + 16);   // K = k0 + 16 + 8h .. +7
  return r.v;
}

// B-matrix fragment (32x16 bf16): lane holds column n = nb+(lane&15),
// K = k0+8h..+7 then k0+16+8h..+7 (h = lane>>4). W row-major [out,in]: a W row IS a B column.
__device__ __forceinline__ v16bf load_b_frag(const unsigned short* p) {
  FragBF r;
  r.u[0] = *(const uint4*)p;
  r.u[1] = *(const uint4*)(p + 16);
  return r.v;
}

// C[M,N] = A[M,Kd] @ W[N,Kd]^T (+ bias[n]) (+ Cadd elementwise)
// Block 256 threads = 8 waves (2M x 4N), tile 128x256; wave tile 64x64 (4x4 16x16 frags).
// Branch-free bf16 inner loop: 16 global_load_b128 : 16 v_wmma per K-step of 32.
__global__ __launch_bounds__(256)
void wmma_gemm(const unsigned short* __restrict__ A, int lda,
               const unsigned short* __restrict__ W, int ldw,
               float* __restrict__ C, int ldc,
               const float* __restrict__ Cadd,
               const float* __restrict__ bias,
               int Kd) {
  const int lane = threadIdx.x & 31;
  const int wid  = threadIdx.x >> 5;
  const int wm   = wid >> 2;          // 0..1
  const int wn   = wid & 3;           // 0..3
  const int mb   = blockIdx.y * 128 + wm * 64;
  const int nb   = blockIdx.x * 256 + wn * 64;
  const int l15  = lane & 15;
  const int h    = lane >> 4;

  // lane-resolved base pointers (K advances by 32 each iteration)
  const unsigned short* ap = A + (size_t)(mb + l15) * lda + 8 * h;
  const unsigned short* bp = W + (size_t)(nb + l15) * ldw + 8 * h;
  const size_t a_row16 = (size_t)16 * lda;   // 16 rows of A
  const size_t b_col16 = (size_t)16 * ldw;   // 16 cols of C = 16 rows of W

  v8f acc[4][4] = {};
  for (int k0 = 0; k0 < Kd; k0 += 32) {
    v16bf a0 = load_a_frag(ap + k0);
    v16bf a1 = load_a_frag(ap + a_row16 + k0);
    v16bf a2 = load_a_frag(ap + 2 * a_row16 + k0);
    v16bf a3 = load_a_frag(ap + 3 * a_row16 + k0);
    v16bf b0 = load_b_frag(bp + k0);
    v16bf b1 = load_b_frag(bp + b_col16 + k0);
    v16bf b2 = load_b_frag(bp + 2 * b_col16 + k0);
    v16bf b3 = load_b_frag(bp + 3 * b_col16 + k0);
#define WM(i, j, av, bv) \
    acc[i][j] = __builtin_amdgcn_wmma_f32_16x16x32_bf16(false, av, false, bv, \
                                                        (short)0, acc[i][j], false, false)
    WM(0, 0, a0, b0); WM(0, 1, a0, b1); WM(0, 2, a0, b2); WM(0, 3, a0, b3);
    WM(1, 0, a1, b0); WM(1, 1, a1, b1); WM(1, 2, a1, b2); WM(1, 3, a1, b3);
    WM(2, 0, a2, b0); WM(2, 1, a2, b1); WM(2, 2, a2, b2); WM(2, 3, a2, b3);
    WM(3, 0, a3, b0); WM(3, 1, a3, b1); WM(3, 2, a3, b2); WM(3, 3, a3, b3);
#undef WM
  }

  // C/D frag layout: lane -> (M = r + 8*(lane>>4), N = lane&15) per 16x16 frag
#pragma unroll
  for (int i = 0; i < 4; ++i) {
#pragma unroll
    for (int j = 0; j < 4; ++j) {
      const int col = nb + 16 * j + l15;
      const float bv = bias ? bias[col] : 0.0f;
#pragma unroll
      for (int r = 0; r < 8; ++r) {
        const int row = mb + 16 * i + r + 8 * h;
        const size_t idx = (size_t)row * ldc + col;
        float v = acc[i][j][r] + bv;
        if (Cadd) v += Cadd[idx];
        C[idx] = v;
      }
    }
  }
}

// ---- pack / convert kernels -------------------------------------------------

// W2[n][0:1024] = W_b part, W2[n][1024:2048] = W_c part; Wq[n][:] = W_q part.
// n<1024 -> gate rows, n>=1024 -> proj rows.
__global__ void pack_weights(const float* __restrict__ gw, const float* __restrict__ pw,
                             unsigned short* __restrict__ W2, unsigned short* __restrict__ Wq) {
  int idx = blockIdx.x * blockDim.x + threadIdx.x;
  if (idx >= 2048 * 3072) return;
  int n = idx / 3072;
  int c = idx - n * 3072;
  float v = (n < 1024) ? gw[(size_t)n * 3072 + c] : pw[(size_t)(n - 1024) * 3072 + c];
  unsigned short hv = f2bf(v);
  if (c < 1024)            W2[(size_t)n * 2048 + c] = hv;
  else if (c < 2048)       Wq[(size_t)n * 1024 + (c - 1024)] = hv;
  else                     W2[(size_t)n * 2048 + 1024 + (c - 2048)] = hv;
}

__global__ void cvt_bf16(const float* __restrict__ in, unsigned short* __restrict__ out, int n) {
  int i = blockIdx.x * blockDim.x + threadIdx.x;
  if (i < n) out[i] = f2bf(in[i]);
}

__global__ void pack_bias2(const float* __restrict__ gb, const float* __restrict__ pb,
                           float* __restrict__ bias2) {
  int i = blockIdx.x * blockDim.x + threadIdx.x;
  if (i < 2048) bias2[i] = (i < 1024) ? gb[i] : pb[i - 1024];
}

// Seed scan state: b = 0, abuf = [bf16(0) | bf16(c_0)]
__global__ void init_a(const float* __restrict__ chunks,
                       float* __restrict__ b, unsigned short* __restrict__ abf) {
  int idx = blockIdx.x * blockDim.x + threadIdx.x;   // over 2048*1024
  int row = idx >> 10;
  int col = idx & 1023;
  b[idx] = 0.0f;
  unsigned short* arow = abf + (size_t)row * 2048;
  arow[col] = 0;
  arow[1024 + col] = f2bf(chunks[(size_t)row * 32768 + col]);   // k = 0
}

__global__ void cvt_fc1(const float* __restrict__ w, unsigned short* __restrict__ out) {
  int i = blockIdx.x * blockDim.x + threadIdx.x;
  if (i >= 1024 * FPAD) return;
  int n = i / FPAD;
  int c = i - n * FPAD;
  out[i] = (c < FREAL) ? f2bf(w[(size_t)n * FREAL + c]) : (unsigned short)0;
}

// ---- per-step elementwise gated update -------------------------------------
// gp holds gate/proj pre-activations incl. q-contribution & bias.
// Also stages next-step A matrix: abuf = [bf16(b_new) | bf16(c_{k+1})].
__global__ __launch_bounds__(256)
void belief_update(const float* __restrict__ gp, const float* __restrict__ sims,
                   const float* __restrict__ chunks, int k,
                   float* __restrict__ b, unsigned short* __restrict__ abf) {
  int idx = blockIdx.x * 256 + threadIdx.x;
  int row = idx >> 10;
  int col = idx & 1023;
  float s = sims[row * 32 + k];
  float w = fminf(fmaxf(s, 0.0f), 1.0f);
  const float* gr = gp + (size_t)row * 2048;
  float zg = gr[col];
  float zh = gr[1024 + col];
  float g  = w / (1.0f + __expf(-zg));
  float h  = tanhf(zh);
  float bo = b[idx];
  float bn = (1.0f - g) * bo + g * h;
  b[idx] = bn;
  unsigned short* arow = abf + (size_t)row * 2048;
  arow[col] = f2bf(bn);
  if (k + 1 < 32)
    arow[1024 + col] = f2bf(chunks[((size_t)row * 32 + (k + 1)) * 1024 + col]);
}

// ---- head: feats + LayerNorm (fused, one block per row) --------------------
__global__ __launch_bounds__(256)
void feats_ln(const float* __restrict__ b, const float* __restrict__ q,
              const float* __restrict__ sims,
              const float* __restrict__ ln_g, const float* __restrict__ ln_b,
              unsigned short* __restrict__ fn) {
  __shared__ float sf[FREAL];
  __shared__ float rs[8], rq[8];
  __shared__ float stats[4];
  const int row = blockIdx.x;
  const int tid = threadIdx.x;
  if (tid < 32) {
    float v = sims[row * 32 + tid];
    float mx = v, sm = v;
#pragma unroll
    for (int o = 16; o > 0; o >>= 1) {
      mx = fmaxf(mx, __shfl_xor(mx, o));
      sm += __shfl_xor(sm, o);
    }
    if (tid == 0) { stats[0] = mx; stats[1] = sm * (1.0f / 32.0f); }
  }
  __syncthreads();
  const float* brow = b + (size_t)row * 1024;
  const float* qrow = q + (size_t)row * 1024;
  float ps = 0.0f, pq = 0.0f;
  for (int j = tid; j < FREAL; j += 256) {
    float v;
    if (j < 1024)       v = brow[j];
    else if (j < 2048)  v = qrow[j - 1024];
    else if (j < 3072)  v = fabsf(brow[j - 2048] - qrow[j - 2048]);
    else if (j < 4096)  v = brow[j - 3072] * qrow[j - 3072];
    else                v = stats[j - 4096];
    sf[j] = v; ps += v; pq += v * v;
  }
#pragma unroll
  for (int o = 16; o > 0; o >>= 1) { ps += __shfl_xor(ps, o); pq += __shfl_xor(pq, o); }
  if ((tid & 31) == 0) { rs[tid >> 5] = ps; rq[tid >> 5] = pq; }
  __syncthreads();
  if (tid == 0) {
    float ts = 0.0f, tq = 0.0f;
#pragma unroll
    for (int i = 0; i < 8; ++i) { ts += rs[i]; tq += rq[i]; }
    float mu  = ts * (1.0f / FREAL);
    float var = tq * (1.0f / FREAL) - mu * mu;
    stats[2] = mu;
    stats[3] = rsqrtf(var + 1e-5f);
  }
  __syncthreads();
  const float mu = stats[2], rstd = stats[3];
  unsigned short* frow = fn + (size_t)row * FPAD;
  for (int j = tid; j < FPAD; j += 256) {
    float o = 0.0f;
    if (j < FREAL) o = (sf[j] - mu) * rstd * ln_g[j] + ln_b[j];
    frow[j] = f2bf(o);
  }
}

// ---- head: exact GELU + fc2 dot reduce -------------------------------------
__global__ __launch_bounds__(256)
void head_out(const float* __restrict__ fc1out,
              const float* __restrict__ fc2_w, const float* __restrict__ fc2_b,
              float* __restrict__ out) {
  __shared__ float rs[8];
  const int row = blockIdx.x;
  const int tid = threadIdx.x;
  const float* hrow = fc1out + (size_t)row * 1024;
  float p = 0.0f;
  for (int j = tid; j < 1024; j += 256) {
    float x = hrow[j];
    float g = 0.5f * x * (1.0f + erff(x * 0.70710678118654752f));
    p += g * fc2_w[j];
  }
#pragma unroll
  for (int o = 16; o > 0; o >>= 1) p += __shfl_xor(p, o);
  if ((tid & 31) == 0) rs[tid >> 5] = p;
  __syncthreads();
  if (tid == 0) {
    float t = 0.0f;
#pragma unroll
    for (int i = 0; i < 8; ++i) t += rs[i];
    out[row] = t + fc2_b[0];
  }
}

extern "C" void kernel_launch(void* const* d_in, const int* in_sizes, int n_in,
                              void* d_out, int out_size, void* d_ws, size_t ws_size,
                              hipStream_t stream) {
  (void)in_sizes; (void)n_in; (void)out_size; (void)ws_size;
  const float* q      = (const float*)d_in[0];
  const float* chunks = (const float*)d_in[1];
  const float* sims   = (const float*)d_in[2];
  const float* gate_w = (const float*)d_in[3];
  const float* gate_b = (const float*)d_in[4];
  const float* proj_w = (const float*)d_in[5];
  const float* proj_b = (const float*)d_in[6];
  const float* ln_g   = (const float*)d_in[7];
  const float* ln_b   = (const float*)d_in[8];
  const float* fc1_w  = (const float*)d_in[9];
  const float* fc1_b  = (const float*)d_in[10];
  const float* fc2_w  = (const float*)d_in[11];
  const float* fc2_b  = (const float*)d_in[12];
  float* out = (float*)d_out;

  char* ws = (char*)d_ws;
  unsigned short* W2   = (unsigned short*)(ws + 0);          //  8.0 MiB bf16 [2048][2048]
  unsigned short* Wq   = (unsigned short*)(ws + 8388608);    //  4.0 MiB bf16 [2048][1024]
  unsigned short* qbf  = (unsigned short*)(ws + 12582912);   //  4.0 MiB bf16 [2048][1024]
  float* qc            = (float*)(ws + 16777216);            // 16.0 MiB f32  [2048][2048]
  float* bbuf          = (float*)(ws + 33554432);            //  8.0 MiB f32  [2048][1024]
  unsigned short* abuf = (unsigned short*)(ws + 41943040);   //  8.0 MiB bf16 [2048][2048]
  float* gp            = (float*)(ws + 50331648);            // 16.0 MiB f32  [2048][2048]
  float* bias2         = (float*)(ws + 67108864);            //  8 KiB  f32  [2048]
  // Reuse of dead regions after the scan loop:
  unsigned short* fc1wp = (unsigned short*)(ws + 0);         // over W2+Wq (dead)
  unsigned short* fn    = (unsigned short*)(ws + 12582912);  // over qbf+qc (dead)
  float* fc1out         = (float*)(ws + 50331648);           // over gp (dead)
  // total workspace footprint: ~64 MiB

  dim3 blk(256);
  pack_weights<<<(2048 * 3072) / 256, blk, 0, stream>>>(gate_w, proj_w, W2, Wq);
  cvt_bf16<<<(2048 * 1024) / 256, blk, 0, stream>>>(q, qbf, 2048 * 1024);
  pack_bias2<<<8, blk, 0, stream>>>(gate_b, proj_b, bias2);
  init_a<<<(2048 * 1024) / 256, blk, 0, stream>>>(chunks, bbuf, abuf);

  // qc = q @ [Wq_gate; Wq_proj]^T + [gate_b; proj_b]  (step-invariant)
  wmma_gemm<<<dim3(8, 16), blk, 0, stream>>>(qbf, 1024, Wq, 1024,
                                             qc, 2048, nullptr, bias2, 1024);

  // Sequential scan: one fused gate+proj GEMM per step over A = [b | c_k] (bf16, contiguous)
  for (int k = 0; k < 32; ++k) {
    wmma_gemm<<<dim3(8, 16), blk, 0, stream>>>(abuf, 2048, W2, 2048,
                                               gp, 2048, qc, nullptr, 2048);
    belief_update<<<(2048 * 1024) / 256, blk, 0, stream>>>(gp, sims, chunks, k, bbuf, abuf);
  }

  // Head
  cvt_fc1<<<(1024 * FPAD) / 256, blk, 0, stream>>>(fc1_w, fc1wp);
  feats_ln<<<2048, blk, 0, stream>>>(bbuf, q, sims, ln_g, ln_b, fn);
  wmma_gemm<<<dim3(4, 16), blk, 0, stream>>>(fn, FPAD, fc1wp, FPAD,
                                             fc1out, 1024, nullptr, fc1_b, FPAD);
  head_out<<<2048, blk, 0, stream>>>(fc1out, fc2_w, fc2_b, out);
}
